// ConfidenceGNNFusion_76802605187252
// MI455X (gfx1250) — matrix operand order repaired and checked by reading, not compiled, not run on hardware
//
#include <hip/hip_runtime.h>
#include <hip/hip_bf16.h>

#define NN   512
#define CIN  64
#define HH   32
#define WW   32
#define HID  128
#define HEADS 4
#define EE   16384
#define ET   (EE + NN)
#define HWSZ (HH * WW)

typedef __attribute__((ext_vector_type(16))) __bf16 v16bf;
typedef __attribute__((ext_vector_type(8)))  float  v8f;
typedef __attribute__((ext_vector_type(8)))  unsigned int v8u;

static __device__ __forceinline__ unsigned short f2bf(float f) {
  unsigned u = __float_as_uint(f);
  unsigned r = u + 0x7FFFu + ((u >> 16) & 1u);   // round-to-nearest-even
  return (unsigned short)(r >> 16);
}
static __device__ __forceinline__ float bf2f(unsigned h) {
  return __uint_as_float(h << 16);
}
static __device__ __forceinline__ unsigned pack2(float a, float b) {
  return (unsigned)f2bf(a) | ((unsigned)f2bf(b) << 16);
}
static __device__ __forceinline__ v16bf frag_from(v8u a) {
  return __builtin_bit_cast(v16bf, a);
}
static __device__ __forceinline__ v8f wmma_bf16(v16bf A, v16bf B, v8f C) {
  return __builtin_amdgcn_wmma_f32_16x16x32_bf16(false, A, false, B, (short)0, C,
                                                 false, false);
}
// CDNA5 async global->LDS DMA (ASYNCcnt-tracked). lds_off is the LDS byte
// offset (single __shared__ object per kernel => allocation starts at 0).
static __device__ __forceinline__ void async_ld_b128(unsigned lds_off,
                                                     const void* gaddr) {
  asm volatile("global_load_async_to_lds_b128 %0, %1, off"
               :: "v"(lds_off), "v"(gaddr) : "memory");
}
static __device__ __forceinline__ void async_wait0() {
  asm volatile("s_wait_asynccnt 0x0" ::: "memory");
}
// monotonic float<->uint encoding for atomicMax on floats
static __device__ __forceinline__ unsigned fenc(float f) {
  unsigned u = __float_as_uint(f);
  return (u & 0x80000000u) ? ~u : (u | 0x80000000u);
}
static __device__ __forceinline__ float fdec(unsigned e) {
  unsigned u = (e & 0x80000000u) ? (e ^ 0x80000000u) : ~e;
  return __uint_as_float(u);
}

// -------- weight packing: OIHW f32 -> [p][ci/2][co] packed bf16 pairs ------
__global__ void pack_weights(const float* __restrict__ w, unsigned* __restrict__ wp,
                             int co_n, int ci_n, int khw) {
  int cp_n = ci_n >> 1;
  int total = khw * cp_n * co_n;
  int idx = blockIdx.x * blockDim.x + threadIdx.x;
  if (idx >= total) return;
  int co = idx % co_n;
  int t  = idx / co_n;
  int cp = t % cp_n;
  int p  = t / cp_n;
  float a = w[(size_t)(co * ci_n + 2 * cp) * khw + p];
  float b = w[(size_t)(co * ci_n + 2 * cp + 1) * khw + p];
  wp[idx] = pack2(a, b);
}

// -------- feature encoder: 1x1 conv (64->128) + BN + ReLU + confidence -----
// writes packed bf16 activations: [N][HW][64cp] dwords (16-bit halves per ch)
__global__ void __launch_bounds__(128)
fe_conv1x1(const float* __restrict__ x,      // [N][64][HW] f32 NCHW
           const unsigned* __restrict__ wp,  // [32][128]
           const float* __restrict__ bn_g, const float* __restrict__ conv_b,
           const float* __restrict__ bn_b,
           const float* __restrict__ conf,   // [N][HW]
           unsigned short* __restrict__ out) // [N][HW][128] bf16
{
  __shared__ __align__(16) unsigned sA[16][32];
  int n = blockIdx.y;
  int pix0 = blockIdx.x * 16;
  int t = threadIdx.x;
  for (int i = t; i < 16 * 32; i += 128) {
    int cp = i >> 4, pix = i & 15;
    const float* src = x + (size_t)(n * CIN + 2 * cp) * HWSZ + pix0 + pix;
    sA[pix][cp] = pack2(src[0], src[HWSZ]);
  }
  __syncthreads();
  int wave = t >> 5, lane = t & 31, col = lane & 15;
  bool hi = lane >= 16;
  v8f acc0 = {0, 0, 0, 0, 0, 0, 0, 0};
  v8f acc1 = {0, 0, 0, 0, 0, 0, 0, 0};
  int co0 = wave * 32;
  const uint4* arow = (const uint4*)&sA[col][0];
#pragma unroll
  for (int q = 0; q < 2; ++q) {
    int cp0 = q * 16;
    int abase = cp0 + (hi ? 8 : 0);
    uint4 al = arow[abase >> 2];
    uint4 ah = arow[(abase + 4) >> 2];
    v8u a; a[0]=al.x; a[1]=al.y; a[2]=al.z; a[3]=al.w;
           a[4]=ah.x; a[5]=ah.y; a[6]=ah.z; a[7]=ah.w;
    v16bf A = frag_from(a);
    int kb = cp0 + (hi ? 8 : 0);
    v8u b0, b1;
#pragma unroll
    for (int j = 0; j < 8; ++j) {
      b0[j] = wp[(kb + j) * HID + co0 + col];
      b1[j] = wp[(kb + j) * HID + co0 + 16 + col];
    }
    acc0 = wmma_bf16(A, frag_from(b0), acc0);
    acc1 = wmma_bf16(A, frag_from(b1), acc1);
  }
  float inv = rsqrtf(1.0f + 1e-5f);
  int c0 = co0 + col, c1 = co0 + 16 + col;
  float sc0 = bn_g[c0] * inv, sc1 = bn_g[c1] * inv;
  float eb0 = conv_b[c0] * sc0 + bn_b[c0];
  float eb1 = conv_b[c1] * sc1 + bn_b[c1];
#pragma unroll
  for (int r = 0; r < 8; ++r) {
    int m = r + (hi ? 8 : 0);
    float cf = conf[(size_t)n * HWSZ + pix0 + m];
    size_t row = (size_t)(n * HWSZ + pix0 + m) * HID;
    out[row + c0] = f2bf(fmaxf(acc0[r] * sc0 + eb0, 0.0f) * cf);
    out[row + c1] = f2bf(fmaxf(acc1[r] * sc1 + eb1, 0.0f) * cf);
  }
}

// -------- spatial processor: 3x3 conv (128->128) + BN + ReLU ---------------
// packed bf16 in [N][HW][64cp] dwords, packed bf16 out; async LDS staging
__global__ void __launch_bounds__(128)
conv3x3_bn_relu(const unsigned* __restrict__ in, // [N][HW][64] dwords
                const unsigned* __restrict__ wp, // [9][64][128]
                const float* __restrict__ conv_b, const float* __restrict__ bn_g,
                const float* __restrict__ bn_b,
                unsigned short* __restrict__ out)// [N][HW][128] bf16
{
  __shared__ __align__(16) unsigned sA[3][18][64];
  int n = blockIdx.z, h0 = blockIdx.y, w0 = blockIdx.x * 16;
  int t = threadIdx.x;
  // 54 rows of 64 dwords; 16 b128 chunks per row; OOB halo rows -> DS zeros,
  // in-range rows -> async DMA straight from global (identical layout).
  for (int i = t; i < 54 * 16; i += 128) {
    int cq = i & 15, row = i >> 4;
    int wc = row % 18, r = row / 18;
    int hh = h0 + r - 1, ww = w0 + wc - 1;
    unsigned lds_off = (unsigned)((row * 64 + cq * 4) * 4);
    if (hh >= 0 && hh < HH && ww >= 0 && ww < WW) {
      const unsigned* g = in + ((size_t)(n * HWSZ + hh * WW + ww) * 64 + cq * 4);
      async_ld_b128(lds_off, g);
    } else {
      uint4 z = {0u, 0u, 0u, 0u};
      *(uint4*)((char*)&sA[0][0][0] + lds_off) = z;
    }
  }
  async_wait0();
  __syncthreads();
  int wave = t >> 5, lane = t & 31, col = lane & 15;
  bool hi = lane >= 16;
  v8f acc0 = {0, 0, 0, 0, 0, 0, 0, 0};
  v8f acc1 = {0, 0, 0, 0, 0, 0, 0, 0};
  int co0 = wave * 32;
  for (int p = 0; p < 9; ++p) {
    int r = p / 3, kw = p % 3;
    const uint4* arow = (const uint4*)&sA[r][col + kw][0];
#pragma unroll
    for (int q = 0; q < 4; ++q) {
      int cp0 = q * 16;
      int abase = cp0 + (hi ? 8 : 0);
      uint4 al = arow[abase >> 2];
      uint4 ah = arow[(abase + 4) >> 2];
      v8u a; a[0]=al.x; a[1]=al.y; a[2]=al.z; a[3]=al.w;
             a[4]=ah.x; a[5]=ah.y; a[6]=ah.z; a[7]=ah.w;
      v16bf A = frag_from(a);
      int kb = p * 64 + cp0 + (hi ? 8 : 0);
      v8u b0, b1;
#pragma unroll
      for (int j = 0; j < 8; ++j) {
        b0[j] = wp[(kb + j) * HID + co0 + col];
        b1[j] = wp[(kb + j) * HID + co0 + 16 + col];
      }
      acc0 = wmma_bf16(A, frag_from(b0), acc0);
      acc1 = wmma_bf16(A, frag_from(b1), acc1);
    }
  }
  float inv = rsqrtf(1.0f + 1e-5f);
  int c0 = co0 + col, c1 = co0 + 16 + col;
  float sc0 = bn_g[c0] * inv, sc1 = bn_g[c1] * inv;
  float eb0 = conv_b[c0] * sc0 + bn_b[c0];
  float eb1 = conv_b[c1] * sc1 + bn_b[c1];
#pragma unroll
  for (int r = 0; r < 8; ++r) {
    int m = r + (hi ? 8 : 0);
    size_t row = (size_t)(n * HWSZ + h0 * WW + w0 + m) * HID;
    out[row + c0] = f2bf(fmaxf(acc0[r] * sc0 + eb0, 0.0f));
    out[row + c1] = f2bf(fmaxf(acc1[r] * sc1 + eb1, 0.0f));
  }
}

// -------- global average pool over packed bf16: -> [N][128] f32 ------------
__global__ void __launch_bounds__(256)
avgpool(const unsigned* __restrict__ xp,    // [N][HW][64] dwords
        float* __restrict__ xnode)
{
  __shared__ float acc[HID];
  int n = blockIdx.x, t = threadIdx.x;
  for (int c = t; c < HID; c += 256) acc[c] = 0.0f;
  __syncthreads();
  const unsigned* base = xp + (size_t)n * HWSZ * 64;
  for (int i = t; i < HWSZ * 64; i += 256) {
    unsigned v = base[i];
    int cp = i & 63;
    atomicAdd(&acc[2 * cp],     bf2f(v & 0xFFFFu));
    atomicAdd(&acc[2 * cp + 1], bf2f(v >> 16));
  }
  __syncthreads();
  for (int c = t; c < HID; c += 256)
    xnode[n * HID + c] = acc[c] * (1.0f / HWSZ);
}

// -------- GAT linear: [512][128] @ [128][512] -> [512][512] via WMMA -------
__global__ void __launch_bounds__(128)
gat_lin_gemm(const float* __restrict__ xin,  // [512][128]
             const float* __restrict__ wlin, // [128][512]
             float* __restrict__ hout)       // [512][512]
{
  __shared__ __align__(16) unsigned sA[16][64];
  int m0 = blockIdx.x * 16, n0 = blockIdx.y * 128;
  int t = threadIdx.x;
  for (int i = t; i < 16 * 64; i += 128) {
    int m = i >> 6, cp = i & 63;
    const float* src = xin + (size_t)(m0 + m) * HID + 2 * cp;
    sA[m][cp] = pack2(src[0], src[1]);
  }
  __syncthreads();
  int wave = t >> 5, lane = t & 31, col = lane & 15;
  bool hi = lane >= 16;
  v8f acc0 = {0, 0, 0, 0, 0, 0, 0, 0};
  v8f acc1 = {0, 0, 0, 0, 0, 0, 0, 0};
  int co0 = n0 + wave * 32;
  const uint4* arow = (const uint4*)&sA[col][0];
#pragma unroll
  for (int q = 0; q < 4; ++q) {
    int cp0 = q * 16;
    int abase = cp0 + (hi ? 8 : 0);
    uint4 al = arow[abase >> 2];
    uint4 ah = arow[(abase + 4) >> 2];
    v8u a; a[0]=al.x; a[1]=al.y; a[2]=al.z; a[3]=al.w;
           a[4]=ah.x; a[5]=ah.y; a[6]=ah.z; a[7]=ah.w;
    v16bf A = frag_from(a);
    int kb = cp0 + (hi ? 8 : 0);
    v8u b0, b1;
#pragma unroll
    for (int j = 0; j < 8; ++j) {
      int k = 2 * (kb + j);
      b0[j] = pack2(wlin[(size_t)k * 512 + co0 + col],
                    wlin[(size_t)(k + 1) * 512 + co0 + col]);
      b1[j] = pack2(wlin[(size_t)k * 512 + co0 + 16 + col],
                    wlin[(size_t)(k + 1) * 512 + co0 + 16 + col]);
    }
    acc0 = wmma_bf16(A, frag_from(b0), acc0);
    acc1 = wmma_bf16(A, frag_from(b1), acc1);
  }
#pragma unroll
  for (int r = 0; r < 8; ++r) {
    int m = r + (hi ? 8 : 0);
    hout[(size_t)(m0 + m) * 512 + co0 + col] = acc0[r];
    hout[(size_t)(m0 + m) * 512 + co0 + 16 + col] = acc1[r];
  }
}

// -------- attention scores per (node, head) --------------------------------
__global__ void __launch_bounds__(128)
gat_scores(const float* __restrict__ h, const float* __restrict__ att_s,
           const float* __restrict__ att_d,
           float* __restrict__ a_s, float* __restrict__ a_d)
{
  int n = blockIdx.x;
  int t = threadIdx.x;
  int head = t >> 5, lane = t & 31;
  float ss = 0.0f, sd = 0.0f;
  for (int d = lane; d < HID; d += 32) {
    float hv = h[(size_t)n * 512 + head * HID + d];
    ss += hv * att_s[head * HID + d];
    sd += hv * att_d[head * HID + d];
  }
#pragma unroll
  for (int o = 16; o > 0; o >>= 1) {
    ss += __shfl_down(ss, o, 32);
    sd += __shfl_down(sd, o, 32);
  }
  if (lane == 0) { a_s[n * 4 + head] = ss; a_d[n * 4 + head] = sd; }
}

static __device__ __forceinline__ void edge_sd(const long long* ei, int e,
                                               int& s, int& d) {
  if (e < EE) { s = (int)ei[e]; d = (int)ei[EE + e]; }
  else        { s = d = e - EE; }
}

__global__ void edge_max(const long long* __restrict__ ei,
                         const float* __restrict__ as, const float* __restrict__ ad,
                         unsigned* __restrict__ menc)
{
  int e = blockIdx.x * blockDim.x + threadIdx.x;
  if (e >= ET) return;
  int s, d; edge_sd(ei, e, s, d);
#pragma unroll
  for (int hd = 0; hd < HEADS; ++hd) {
    float v = as[s * 4 + hd] + ad[d * 4 + hd];
    v = v > 0.0f ? v : 0.2f * v;
    atomicMax(menc + d * 4 + hd, fenc(v));
  }
}

__global__ void edge_sum(const long long* __restrict__ ei,
                         const float* __restrict__ as, const float* __restrict__ ad,
                         const unsigned* __restrict__ menc,
                         float* __restrict__ denom)
{
  int e = blockIdx.x * blockDim.x + threadIdx.x;
  if (e >= ET) return;
  int s, d; edge_sd(ei, e, s, d);
#pragma unroll
  for (int hd = 0; hd < HEADS; ++hd) {
    float v = as[s * 4 + hd] + ad[d * 4 + hd];
    v = v > 0.0f ? v : 0.2f * v;
    atomicAdd(denom + d * 4 + hd, expf(v - fdec(menc[d * 4 + hd])));
  }
}

__global__ void __launch_bounds__(128)
edge_aggr(const long long* __restrict__ ei,
          const float* __restrict__ as, const float* __restrict__ ad,
          const unsigned* __restrict__ menc, const float* __restrict__ denom,
          const float* __restrict__ h, float* __restrict__ aggr)
{
  int e = blockIdx.x;
  int td = threadIdx.x;   // 0..127 = feature dim
  int s, d; edge_sd(ei, e, s, d);
#pragma unroll
  for (int hd = 0; hd < HEADS; ++hd) {
    float v = as[s * 4 + hd] + ad[d * 4 + hd];
    v = v > 0.0f ? v : 0.2f * v;
    float alpha = expf(v - fdec(menc[d * 4 + hd])) / (denom[d * 4 + hd] + 1e-16f);
    atomicAdd(aggr + (size_t)(d * 4 + hd) * HID + td,
              alpha * h[(size_t)s * 512 + hd * HID + td]);
  }
}

__global__ void __launch_bounds__(128)
gat_finalize(const float* __restrict__ aggr, const float* __restrict__ bias,
             float* __restrict__ xg)
{
  int n = blockIdx.x, c = threadIdx.x;
  float s = 0.0f;
#pragma unroll
  for (int hd = 0; hd < HEADS; ++hd)
    s += aggr[(size_t)(n * 4 + hd) * HID + c];
  xg[n * HID + c] = fmaxf(s * 0.25f + bias[c], 0.0f);
}

// -------- per-(n,co) bias from GNN output for the output projector ---------
__global__ void __launch_bounds__(64)
op_bias(const float* __restrict__ op_w, const float* __restrict__ op_b,
        const float* __restrict__ xg, float* __restrict__ nb)
{
  int n = blockIdx.x, co = threadIdx.x;   // 64 threads
  float s = op_b[co];
  for (int ci = 0; ci < HID; ++ci)
    s += op_w[co * HID + ci] * xg[n * HID + ci];
  nb[n * 64 + co] = s;
}

// -------- output projector: 1x1 conv (128->64) + per-node bias -------------
__global__ void __launch_bounds__(128)
op_conv1x1(const unsigned* __restrict__ xp,  // [N][HW][64] dwords packed bf16
           const unsigned* __restrict__ wp,  // [64][64]
           const float* __restrict__ nb,     // [N][64]
           float* __restrict__ out)          // [N][64][HW] f32
{
  __shared__ __align__(16) unsigned sA[16][64];
  int n = blockIdx.y, pix0 = blockIdx.x * 16;
  int t = threadIdx.x;
  // direct async DMA: LDS tile is a contiguous copy of 16 pixel rows
  for (int i = t; i < 16 * 16; i += 128) {
    int cq = i & 15, pix = i >> 4;
    unsigned lds_off = (unsigned)((pix * 64 + cq * 4) * 4);
    const unsigned* g = xp + ((size_t)(n * HWSZ + pix0 + pix) * 64 + cq * 4);
    async_ld_b128(lds_off, g);
  }
  async_wait0();
  __syncthreads();
  int wave = t >> 5, lane = t & 31, col = lane & 15;
  bool hi = lane >= 16;
  v8f acc = {0, 0, 0, 0, 0, 0, 0, 0};
  int co0 = wave * 16;
  const uint4* arow = (const uint4*)&sA[col][0];
#pragma unroll
  for (int q = 0; q < 4; ++q) {
    int cp0 = q * 16;
    int abase = cp0 + (hi ? 8 : 0);
    uint4 al = arow[abase >> 2];
    uint4 ah = arow[(abase + 4) >> 2];
    v8u a; a[0]=al.x; a[1]=al.y; a[2]=al.z; a[3]=al.w;
           a[4]=ah.x; a[5]=ah.y; a[6]=ah.z; a[7]=ah.w;
    v16bf A = frag_from(a);
    int kb = cp0 + (hi ? 8 : 0);
    v8u b;
#pragma unroll
    for (int j = 0; j < 8; ++j)
      b[j] = wp[(kb + j) * 64 + co0 + col];
    acc = wmma_bf16(A, frag_from(b), acc);
  }
  float base = nb[n * 64 + co0 + col];
#pragma unroll
  for (int r = 0; r < 8; ++r) {
    int m = r + (hi ? 8 : 0);
    out[(size_t)(n * 64 + co0 + col) * HWSZ + pix0 + m] = acc[r] + base;
  }
}

extern "C" void kernel_launch(void* const* d_in, const int* in_sizes, int n_in,
                              void* d_out, int out_size, void* d_ws, size_t ws_size,
                              hipStream_t stream) {
  const float*     x      = (const float*)d_in[0];
  const long long* ei     = (const long long*)d_in[1];
  const float*     conf   = (const float*)d_in[2];
  const float*     fe_w   = (const float*)d_in[3];
  const float*     fe_b   = (const float*)d_in[4];
  const float*     fe_g   = (const float*)d_in[5];
  const float*     fe_be  = (const float*)d_in[6];
  const float*     sp_w1  = (const float*)d_in[7];
  const float*     sp_b1  = (const float*)d_in[8];
  const float*     sp_g1  = (const float*)d_in[9];
  const float*     sp_be1 = (const float*)d_in[10];
  const float*     sp_w2  = (const float*)d_in[11];
  const float*     sp_b2  = (const float*)d_in[12];
  const float*     sp_g2  = (const float*)d_in[13];
  const float*     sp_be2 = (const float*)d_in[14];
  const float*     g1_lin = (const float*)d_in[15];
  const float*     g1_as  = (const float*)d_in[16];
  const float*     g1_ad  = (const float*)d_in[17];
  const float*     g1_b   = (const float*)d_in[18];
  const float*     g2_lin = (const float*)d_in[19];
  const float*     g2_as  = (const float*)d_in[20];
  const float*     g2_ad  = (const float*)d_in[21];
  const float*     g2_b   = (const float*)d_in[22];
  const float*     op_w   = (const float*)d_in[23];
  const float*     op_b   = (const float*)d_in[24];

  char* ws = (char*)d_ws;
  size_t off = 0;
  auto alloc = [&](size_t bytes) -> void* {
    void* p = ws + off;
    off = (off + bytes + 255) & ~(size_t)255;
    return p;
  };
  const size_t packed_bytes = (size_t)NN * HWSZ * 64 * sizeof(unsigned); // bf16 acts
  unsigned* packA  = (unsigned*)alloc(packed_bytes);  // xe, then xp2 (final)
  unsigned* packB  = (unsigned*)alloc(packed_bytes);  // xp1
  unsigned* wp_fe  = (unsigned*)alloc(32 * 128 * 4);
  unsigned* wp_s1  = (unsigned*)alloc(9 * 64 * 128 * 4);
  unsigned* wp_s2  = (unsigned*)alloc(9 * 64 * 128 * 4);
  unsigned* wp_op  = (unsigned*)alloc(64 * 64 * 4);
  float*    xnode  = (float*)alloc(NN * HID * 4);
  float*    hbuf   = (float*)alloc(NN * 512 * 4);
  float*    a_s    = (float*)alloc(NN * 4 * 4);
  float*    a_d    = (float*)alloc(NN * 4 * 4);
  unsigned* menc   = (unsigned*)alloc(NN * 4 * 4);
  float*    denom  = (float*)alloc(NN * 4 * 4);
  float*    aggr   = (float*)alloc((size_t)NN * 4 * HID * 4);
  float*    xg1    = (float*)alloc(NN * HID * 4);
  float*    xg2    = (float*)alloc(NN * HID * 4);
  float*    nb     = (float*)alloc(NN * 64 * 4);

  // weight packing (tiny)
  pack_weights<<<(32 * 128 + 255) / 256, 256, 0, stream>>>(fe_w, wp_fe, 128, 64, 1);
  pack_weights<<<(9 * 64 * 128 + 255) / 256, 256, 0, stream>>>(sp_w1, wp_s1, 128, 128, 9);
  pack_weights<<<(9 * 64 * 128 + 255) / 256, 256, 0, stream>>>(sp_w2, wp_s2, 128, 128, 9);
  pack_weights<<<(64 * 64 + 255) / 256, 256, 0, stream>>>(op_w, wp_op, 64, 128, 1);

  // feature encoder + confidence -> packed bf16
  fe_conv1x1<<<dim3(HWSZ / 16, NN), 128, 0, stream>>>(
      x, wp_fe, fe_g, fe_b, fe_be, conf, (unsigned short*)packA);
  // spatial processor (async LDS staging)
  conv3x3_bn_relu<<<dim3(WW / 16, HH, NN), 128, 0, stream>>>(
      packA, wp_s1, sp_b1, sp_g1, sp_be1, (unsigned short*)packB);
  conv3x3_bn_relu<<<dim3(WW / 16, HH, NN), 128, 0, stream>>>(
      packB, wp_s2, sp_b2, sp_g2, sp_be2, (unsigned short*)packA);
  // node features
  avgpool<<<NN, 256, 0, stream>>>(packA, xnode);

  auto gat = [&](const float* xin, const float* lin, const float* asw,
                 const float* adw, const float* bias, float* xg_out) {
    gat_lin_gemm<<<dim3(NN / 16, 4), 128, 0, stream>>>(xin, lin, hbuf);
    gat_scores<<<NN, 128, 0, stream>>>(hbuf, asw, adw, a_s, a_d);
    hipMemsetAsync(menc, 0, NN * 4 * 4, stream);
    hipMemsetAsync(denom, 0, NN * 4 * 4, stream);
    hipMemsetAsync(aggr, 0, (size_t)NN * 4 * HID * 4, stream);
    edge_max<<<(ET + 255) / 256, 256, 0, stream>>>(ei, a_s, a_d, menc);
    edge_sum<<<(ET + 255) / 256, 256, 0, stream>>>(ei, a_s, a_d, menc, denom);
    edge_aggr<<<ET, 128, 0, stream>>>(ei, a_s, a_d, menc, denom, hbuf, aggr);
    gat_finalize<<<NN, 128, 0, stream>>>(aggr, bias, xg_out);
  };
  gat(xnode, g1_lin, g1_as, g1_ad, g1_b, xg1);
  gat(xg1, g2_lin, g2_as, g2_ad, g2_b, xg2);

  // fused output projector (broadcast GNN term folded into per-node bias)
  op_bias<<<NN, 64, 0, stream>>>(op_w, op_b, xg2, nb);
  op_conv1x1<<<dim3(HWSZ / 16, NN), 128, 0, stream>>>(packA, wp_op, nb,
                                                      (float*)d_out);
}
